// Attention_16131897163898
// MI455X (gfx1250) — matrix operand order repaired and checked by reading, not compiled, not run on hardware
//
#include <hip/hip_runtime.h>

typedef __attribute__((ext_vector_type(16))) __bf16 v16bf;
typedef __attribute__((ext_vector_type(8)))  __bf16 v8bf;
typedef __attribute__((ext_vector_type(8)))  float  v8f;

union Frag { v16bf v; v8bf h[2]; };

#define B_  2
#define T_  2048
#define D_  1024
#define H_  16
#define DH_ 64

__device__ inline __bf16 f2bf(float f) {
    unsigned u = __float_as_uint(f);
    u += 0x7FFFu + ((u >> 16) & 1u);              // round-to-nearest-even
    unsigned short hs = (unsigned short)(u >> 16);
    union { unsigned short s; __bf16 b; } o;
    o.s = hs;
    return o.b;
}

__device__ inline v8f wmma_bf16(v16bf a, v16bf b, v8f c) {
    return __builtin_amdgcn_wmma_f32_16x16x32_bf16(
        /*neg_a=*/false, a, /*neg_b=*/false, b,
        /*c_mod=*/(short)0, c, /*reuse_a=*/false, /*reuse_b=*/false);
}

__device__ inline v8f v8f_zero() {
    v8f z;
#pragma unroll
    for (int i = 0; i < 8; ++i) z[i] = 0.0f;
    return z;
}

// ---------------- conversion kernels ----------------

__global__ void __launch_bounds__(256)
k_cvt(const float* __restrict__ s, __bf16* __restrict__ d, int n) {
    int i = blockIdx.x * 256 + threadIdx.x;
    if (i < n) d[i] = f2bf(s[i]);
}

// src [K,N] f32 row-major -> dst [N,K] bf16, LDS-tiled 32x32 (coalesced both sides)
__global__ void __launch_bounds__(256)
k_cvt_t(const float* __restrict__ s, __bf16* __restrict__ d, int K, int N) {
    __shared__ __bf16 tile[32][33];
    const int n0 = blockIdx.x * 32;
    const int k0 = blockIdx.y * 32;
    const int tx = threadIdx.x;   // 0..31
    const int ty = threadIdx.y;   // 0..7
#pragma unroll
    for (int i = 0; i < 4; ++i) {
        const int k = k0 + ty + i * 8;
        tile[ty + i * 8][tx] = f2bf(s[(size_t)k * N + n0 + tx]);
    }
    __syncthreads();
#pragma unroll
    for (int i = 0; i < 4; ++i) {
        const int n = n0 + ty + i * 8;
        d[(size_t)n * K + k0 + tx] = tile[tx][ty + i * 8];
    }
}

// ---------------- fused QKV GEMM + split-heads ----------------
// A: xb [M=4096, K=1024] bf16 row-major ; BT: w_attn^T [N=3072, K=1024] bf16
// 32x64 tile per wave: 2 A-frags, 4 B-frags (B reused across both row halves), 8 WMMA/K-step.
__global__ void __launch_bounds__(128)
k_gemm_qkv(const __bf16* __restrict__ A, const __bf16* __restrict__ BT,
           const float* __restrict__ bias,
           __bf16* __restrict__ qb, __bf16* __restrict__ kb, __bf16* __restrict__ vbT,
           float* __restrict__ pK, float* __restrict__ pV)
{
    const int K    = D_;
    const int wave = threadIdx.x >> 5;
    const int lane = threadIdx.x & 31;
    const int half = lane >> 4;
    const int ln   = lane & 15;
    const int m0   = blockIdx.x * 32;
    const int n0   = blockIdx.y * 256 + wave * 64;

    v8f acc[2][4];
#pragma unroll
    for (int r = 0; r < 2; ++r)
#pragma unroll
        for (int i = 0; i < 4; ++i) acc[r][i] = v8f_zero();

    const __bf16* arow0 = A + (size_t)(m0 + ln) * K;
    const __bf16* arow1 = A + (size_t)(m0 + 16 + ln) * K;
    for (int k = 0; k < K; k += 32) {
        __builtin_prefetch(arow0 + k + 512, 0, 3);   // -> global_prefetch_b8
        __builtin_prefetch(arow1 + k + 512, 0, 3);
        Frag a0, a1;
        a0.h[0] = *(const v8bf*)(arow0 + k + half * 16);
        a0.h[1] = *(const v8bf*)(arow0 + k + half * 16 + 8);
        a1.h[0] = *(const v8bf*)(arow1 + k + half * 16);
        a1.h[1] = *(const v8bf*)(arow1 + k + half * 16 + 8);
#pragma unroll
        for (int nt = 0; nt < 4; ++nt) {
            Frag b;
            b.v = *(const v16bf*)(BT + (size_t)(n0 + nt * 16 + ln) * K + k + half * 16);
            acc[0][nt] = wmma_bf16(a0.v, b.v, acc[0][nt]);
            acc[1][nt] = wmma_bf16(a1.v, b.v, acc[1][nt]);
        }
    }

#pragma unroll
    for (int nt = 0; nt < 4; ++nt) {
        const int n     = n0 + nt * 16 + ln;
        const float bs  = bias[n];
        const int which = n >> 10;          // 0=q 1=k 2=v
        const int hd    = n & 1023;
        const int h     = hd >> 6;
        const int d     = hd & 63;
#pragma unroll
        for (int r = 0; r < 2; ++r) {
#pragma unroll
            for (int e = 0; e < 8; ++e) {
                const int m  = m0 + r * 16 + e + 8 * half;
                const int bb = m >> 11;     // batch
                const int t  = m & 2047;
                const float val = acc[r][nt][e] + bs;
                const size_t idx = ((size_t)(bb * H_ + h) * T_ + t) * DH_ + d;
                if (which == 0) {
                    qb[idx] = f2bf(val);
                } else if (which == 1) {
                    kb[idx] = f2bf(val);
                    pK[idx] = val;
                } else {
                    vbT[((size_t)(bb * H_ + h) * DH_ + d) * T_ + t] = f2bf(val);
                    pV[idx] = val;
                }
            }
        }
    }
}

// ---------------- flash attention (causal), one wave per 16-query tile ----------------
__global__ void __launch_bounds__(32)
k_attn(const __bf16* __restrict__ qb, const __bf16* __restrict__ kb,
       const __bf16* __restrict__ vbT, __bf16* __restrict__ ab)
{
    __shared__ __bf16 Pb[16][32];

    const int bh   = blockIdx.y;
    const int q0   = blockIdx.x * 16;
    const int lane = threadIdx.x & 31;
    const int half = lane >> 4;
    const int ln   = lane & 15;

    // Q: two A fragments (dh = 64 = 2 x K32)
    Frag aq[2];
    const __bf16* qrow = qb + ((size_t)bh * T_ + q0 + ln) * DH_;
#pragma unroll
    for (int j = 0; j < 2; ++j) {
        aq[j].h[0] = *(const v8bf*)(qrow + j * 32 + half * 16);
        aq[j].h[1] = *(const v8bf*)(qrow + j * 32 + half * 16 + 8);
    }

    float mrow[8], lrow[8];
#pragma unroll
    for (int e = 0; e < 8; ++e) { mrow[e] = -3.0e38f; lrow[e] = 0.0f; }
    v8f o[4];
#pragma unroll
    for (int c = 0; c < 4; ++c) o[c] = v8f_zero();

    const __bf16* krow = kb + ((size_t)bh * T_ + ln) * DH_;

    const int nblk = (q0 + 47) >> 5;        // key blocks of 32 covering [0, q0+15]
    for (int blk = 0; blk < nblk; ++blk) {
        const int k0 = blk * 32;
        __builtin_prefetch(krow + (size_t)(k0 + 32) * DH_, 0, 3);

        // scores for two 16-key subtiles
        v8f s0 = v8f_zero(), s1 = v8f_zero();
#pragma unroll
        for (int j = 0; j < 2; ++j) {
            Frag b0, b1;
            b0.v = *(const v16bf*)(krow + (size_t)k0 * DH_ + j * 32 + half * 16);
            b1.v = *(const v16bf*)(krow + (size_t)(k0 + 16) * DH_ + j * 32 + half * 16);
            s0 = wmma_bf16(aq[j].v, b0.v, s0);
            s1 = wmma_bf16(aq[j].v, b1.v, s1);
        }

        // scale 1/sqrt(H), causal mask to exactly -1e10, online softmax
        v8f p0, p1;
        float scalef[8];
#pragma unroll
        for (int e = 0; e < 8; ++e) {
            const int qi = q0 + e + 8 * half;
            float v0 = s0[e] * 0.25f;
            float v1 = s1[e] * 0.25f;
            if (k0 + ln > qi)      v0 = -1.0e10f;
            if (k0 + 16 + ln > qi) v1 = -1.0e10f;
            float mx = fmaxf(v0, v1);
            mx = fmaxf(mx, __shfl_xor(mx, 1, 32));
            mx = fmaxf(mx, __shfl_xor(mx, 2, 32));
            mx = fmaxf(mx, __shfl_xor(mx, 4, 32));
            mx = fmaxf(mx, __shfl_xor(mx, 8, 32));
            const float nm = fmaxf(mrow[e], mx);
            const float sc = __expf(mrow[e] - nm);
            const float e0 = __expf(v0 - nm);
            const float e1 = __expf(v1 - nm);
            float rs = e0 + e1;
            rs += __shfl_xor(rs, 1, 32);
            rs += __shfl_xor(rs, 2, 32);
            rs += __shfl_xor(rs, 4, 32);
            rs += __shfl_xor(rs, 8, 32);
            lrow[e]   = lrow[e] * sc + rs;
            mrow[e]   = nm;
            scalef[e] = sc;
            p0[e] = e0;
            p1[e] = e1;
        }
#pragma unroll
        for (int c = 0; c < 4; ++c)
#pragma unroll
            for (int e = 0; e < 8; ++e) o[c][e] *= scalef[e];

        // re-layout P (C-frag) -> A-frag via LDS
        __syncthreads();
#pragma unroll
        for (int e = 0; e < 8; ++e) {
            Pb[e + 8 * half][ln]      = f2bf(p0[e]);
            Pb[e + 8 * half][16 + ln] = f2bf(p1[e]);
        }
        __syncthreads();
        Frag ap;
        ap.h[0] = *(const v8bf*)(&Pb[ln][half * 16]);
        ap.h[1] = *(const v8bf*)(&Pb[ln][half * 16 + 8]);

        // O += P @ V  (V transposed: columns of B are dh dims, K runs over keys)
#pragma unroll
        for (int c = 0; c < 4; ++c) {
            Frag bv;
            bv.v = *(const v16bf*)(vbT + ((size_t)bh * DH_ + c * 16 + ln) * T_ + k0 + half * 16);
            o[c] = wmma_bf16(ap.v, bv.v, o[c]);
        }
    }

    // normalize and store merged-heads activation (bf16 for the proj GEMM)
    const int bb = bh >> 4;
    const int h  = bh & 15;
#pragma unroll
    for (int e = 0; e < 8; ++e) {
        const float inv = 1.0f / lrow[e];
        const int m = q0 + e + 8 * half;
        __bf16* orow = ab + ((size_t)bb * T_ + m) * D_ + h * DH_;
#pragma unroll
        for (int c = 0; c < 4; ++c)
            orow[c * 16 + ln] = f2bf(o[c][e] * inv);
    }
}

// ---------------- output projection GEMM (32x64 per wave) ----------------
__global__ void __launch_bounds__(128)
k_gemm_proj(const __bf16* __restrict__ A, const __bf16* __restrict__ BT,
            const float* __restrict__ bias, float* __restrict__ out)
{
    const int K    = D_;
    const int wave = threadIdx.x >> 5;
    const int lane = threadIdx.x & 31;
    const int half = lane >> 4;
    const int ln   = lane & 15;
    const int m0   = blockIdx.x * 32;
    const int n0   = blockIdx.y * 256 + wave * 64;

    v8f acc[2][4];
#pragma unroll
    for (int r = 0; r < 2; ++r)
#pragma unroll
        for (int i = 0; i < 4; ++i) acc[r][i] = v8f_zero();

    const __bf16* arow0 = A + (size_t)(m0 + ln) * K;
    const __bf16* arow1 = A + (size_t)(m0 + 16 + ln) * K;
    for (int k = 0; k < K; k += 32) {
        __builtin_prefetch(arow0 + k + 512, 0, 3);
        __builtin_prefetch(arow1 + k + 512, 0, 3);
        Frag a0, a1;
        a0.h[0] = *(const v8bf*)(arow0 + k + half * 16);
        a0.h[1] = *(const v8bf*)(arow0 + k + half * 16 + 8);
        a1.h[0] = *(const v8bf*)(arow1 + k + half * 16);
        a1.h[1] = *(const v8bf*)(arow1 + k + half * 16 + 8);
#pragma unroll
        for (int nt = 0; nt < 4; ++nt) {
            Frag b;
            b.v = *(const v16bf*)(BT + (size_t)(n0 + nt * 16 + ln) * K + k + half * 16);
            acc[0][nt] = wmma_bf16(a0.v, b.v, acc[0][nt]);
            acc[1][nt] = wmma_bf16(a1.v, b.v, acc[1][nt]);
        }
    }

#pragma unroll
    for (int nt = 0; nt < 4; ++nt) {
        const int n    = n0 + nt * 16 + ln;
        const float bs = bias[n];
#pragma unroll
        for (int r = 0; r < 2; ++r)
#pragma unroll
            for (int e = 0; e < 8; ++e) {
                const int m = m0 + r * 16 + e + 8 * half;
                out[(size_t)m * D_ + n] = acc[r][nt][e] + bs;
            }
    }
}

// ---------------- launcher ----------------

extern "C" void kernel_launch(void* const* d_in, const int* in_sizes, int n_in,
                              void* d_out, int out_size, void* d_ws, size_t ws_size,
                              hipStream_t stream)
{
    (void)in_sizes; (void)n_in; (void)out_size; (void)ws_size;

    const float* x      = (const float*)d_in[0];
    const float* w_attn = (const float*)d_in[1];
    const float* b_attn = (const float*)d_in[2];
    const float* w_proj = (const float*)d_in[3];
    const float* b_proj = (const float*)d_in[4];
    float* out = (float*)d_out;

    char* p = (char*)d_ws;
    __bf16* xb  = (__bf16*)p; p += (size_t)B_ * T_ * D_ * 2;
    __bf16* waT = (__bf16*)p; p += (size_t)D_ * 3 * D_ * 2;
    __bf16* wpT = (__bf16*)p; p += (size_t)D_ * D_ * 2;
    __bf16* qb  = (__bf16*)p; p += (size_t)B_ * H_ * T_ * DH_ * 2;
    __bf16* kb  = (__bf16*)p; p += (size_t)B_ * H_ * T_ * DH_ * 2;
    __bf16* vbT = (__bf16*)p; p += (size_t)B_ * H_ * T_ * DH_ * 2;
    __bf16* ab  = (__bf16*)p; p += (size_t)B_ * T_ * D_ * 2;

    float* a_out = out;                                   // [B,T,D]
    float* pK    = out + (size_t)B_ * T_ * D_;            // present[0] = k
    float* pV    = pK + (size_t)B_ * H_ * T_ * DH_;       // present[1] = v

    { const int n = B_ * T_ * D_;
      k_cvt<<<(n + 255) / 256, 256, 0, stream>>>(x, xb, n); }

    k_cvt_t<<<dim3((3 * D_) / 32, D_ / 32), dim3(32, 8), 0, stream>>>(w_attn, waT, D_, 3 * D_);
    k_cvt_t<<<dim3(D_ / 32, D_ / 32), dim3(32, 8), 0, stream>>>(w_proj, wpT, D_, D_);

    k_gemm_qkv<<<dim3((B_ * T_) / 32, (3 * D_) / 256), 128, 0, stream>>>(
        xb, waT, b_attn, qb, kb, vbT, pK, pV);

    k_attn<<<dim3(T_ / 16, B_ * H_), 32, 0, stream>>>(qb, kb, vbT, ab);

    k_gemm_proj<<<dim3((B_ * T_) / 32, D_ / 256), 128, 0, stream>>>(
        ab, wpT, b_proj, a_out);
}
